// PowerFlowUnconstrainedSuperNodeGNN_12678743458347
// MI455X (gfx1250) — matrix operand order repaired and checked by reading, not compiled, not run on hardware
//
#include <hip/hip_runtime.h>
#include <hip/hip_bf16.h>

typedef __attribute__((ext_vector_type(16))) _Float16 v16h;
typedef __attribute__((ext_vector_type(8)))  float    v8f;
typedef __attribute__((ext_vector_type(4)))  int      i32x4;

union Frag16 { v16h v; uint4 q[2]; };

// CDNA5 async global->LDS staging (ASYNCcnt-tracked), guarded so a missing
// builtin falls back to the load+ds_store path.
#if defined(__has_builtin)
#if __has_builtin(__builtin_amdgcn_global_load_async_to_lds_b128) && \
    __has_builtin(__builtin_amdgcn_s_wait_asynccnt)
#define ASYNC_LDS 1
#endif
#endif
#ifndef ASYNC_LDS
#define ASYNC_LDS 0
#endif

#if ASYNC_LDS
typedef __attribute__((address_space(1))) i32x4* as1_i32x4;
typedef __attribute__((address_space(3))) i32x4* as3_i32x4;
#endif

// ---------------------------------------------------------------------------
// Pack W_msg (3 x 38 x 32, zero-padded K->64) and W_n (3 x 64 x 32) into
// WMMA B-fragment layout for v_wmma_f32_16x16x32_f16:
//   lane 0-15 hold K=kc*32+{0..7,16..23} (col = lane), lanes 16-31 hold K=kc*32+{8..15,24..31},
//   2 halves per VGPR -> 16 contiguous halves per lane per fragment.
// frag index within a layer = ct*2 + kc  (ct: column tile 0/1, kc: K chunk 0/1)
// ---------------------------------------------------------------------------
__global__ void prep_pack_kernel(const float* __restrict__ W_msg,
                                 const float* __restrict__ W_n,
                                 _Float16* __restrict__ wmsg_pk,
                                 _Float16* __restrict__ wn_pk) {
  const int total = 3 * 2 * 2 * 32 * 16;   // layers * ct * kc * lanes * halves
  for (int t = threadIdx.x; t < total; t += blockDim.x) {
    int j    = t & 15;
    int lane = (t >> 4) & 31;
    int kc   = (t >> 9) & 1;
    int ct   = (t >> 10) & 1;
    int l    = t >> 11;
    int K    = kc * 32 + ((lane < 16) ? 0 : 8) + ((j < 8) ? j : j + 8);
    int col  = (lane & 15) + ct * 16;
    float vm = (K < 38) ? W_msg[(l * 38 + K) * 32 + col] : 0.0f;
    wmsg_pk[t] = (_Float16)vm;
    float vn = W_n[(l * 64 + K) * 32 + col];
    wn_pk[t] = (_Float16)vn;
  }
}

// ---------------------------------------------------------------------------
// init: h0 = P @ W_in + b_in ; V = {1,0}; nfeat row = [V(2)|h(32)|zeros(30)]
// zero agg / pooled / g
// ---------------------------------------------------------------------------
__global__ void init_kernel(const float* __restrict__ P,
                            const float* __restrict__ W_in,
                            const float* __restrict__ b_in,
                            _Float16* __restrict__ nfeat,
                            float* __restrict__ agg,
                            float* __restrict__ pooled,
                            float* __restrict__ g,
                            float* __restrict__ V, int N) {
  if (blockIdx.x == 0 && threadIdx.x < 64) {
    if (threadIdx.x < 32) pooled[threadIdx.x] = 0.0f;
    else                  g[threadIdx.x - 32] = 0.0f;
  }
  int idx = blockIdx.x * blockDim.x + threadIdx.x;
  int n = idx >> 5, c = idx & 31;
  if (n >= N) return;
  float h0 = P[n * 2] * W_in[c] + P[n * 2 + 1] * W_in[32 + c] + b_in[c];
  nfeat[n * 64 + 2 + c] = (_Float16)h0;
  agg[n * 32 + c] = 0.0f;
  if (c < 2) {
    float v = (c == 0) ? 1.0f : 0.0f;
    V[n * 2 + c] = v;
    nfeat[n * 64 + c] = (_Float16)v;
  }
  if (c < 30) nfeat[n * 64 + 34 + c] = (_Float16)0.0f;   // slots 34..63
}

// ---------------------------------------------------------------------------
// Edge messages: per wave, one tile of 32 edges (two 16-row WMMA sub-tiles
// sharing the B fragments and meta).
//   msg_in row (K=64, f16): [V(2)|h(32)|ef(4)|zeros] gathered via sender idx.
//   m = (msg_in @ W_msg + b) * mask ; atomic scatter-add into agg[receiver].
// A-fragment layout (16-bit A 16x32): lane L holds row M=L%16;
//   L<16: K = {b..b+7, b+16..b+23} with b=0 ; L>=16: b=8. (per kc chunk of 32)
// ---------------------------------------------------------------------------
__global__ void __launch_bounds__(256) edge_msg_kernel(
    const _Float16* __restrict__ nfeat,
    const int* __restrict__ senders, const int* __restrict__ receivers,
    const float* __restrict__ ef, const float* __restrict__ emask,
    const _Float16* __restrict__ wpk,   // layer slice: 4 frags * 512 halves
    const float* __restrict__ bmsg,     // layer bias (32)
    float* __restrict__ agg, int ntiles)
{
  __shared__ __align__(16) unsigned int s_rows[8][32 * 32]; // 8 waves*32 rows*128B
  __shared__ int   s_snd[8][32];
  __shared__ int   s_rcv[8][32];
  __shared__ float s_msk[8][32];

  const int wave = threadIdx.x >> 5;
  const int lane = threadIdx.x & 31;
  const int tile = blockIdx.x * 8 + wave;
  const bool valid = tile < ntiles;
  const int base = tile * 32;

  // B fragments (2 column tiles x 2 K chunks), 2x16B contiguous per lane
  Frag16 bfrag[2][2];
  {
    const uint4* wq = (const uint4*)wpk;
    #pragma unroll
    for (int ct = 0; ct < 2; ++ct)
      #pragma unroll
      for (int kc = 0; kc < 2; ++kc) {
        int qi = ((ct * 2 + kc) * 32 + lane) * 2;
        bfrag[ct][kc].q[0] = wq[qi];
        bfrag[ct][kc].q[1] = wq[qi + 1];
      }
  }

  if (valid) {
    int e = base + lane;
    s_snd[wave][lane] = senders[e];
    s_rcv[wave][lane] = receivers[e];
    s_msk[wave][lane] = emask[e];
  }
  __syncthreads();

  if (valid) {
#if ASYNC_LDS
    // 8 async b128 issues stage all 32 rows (4 rows / instruction):
    //   row = i*4 + lane/8 ; 16B chunk = lane%8 ; LDS written directly.
    #pragma unroll
    for (int i = 0; i < 8; ++i) {
      int r = i * 4 + (lane >> 3);
      int s = s_snd[wave][r];
      char* gsrc = (char*)nfeat + (size_t)s * 128 + (lane & 7) * 16;
      unsigned int* ldst = &s_rows[wave][r * 32 + (lane & 7) * 4];
      __builtin_amdgcn_global_load_async_to_lds_b128(
          (as1_i32x4)gsrc, (as3_i32x4)ldst, 0, 0);
    }
    __builtin_amdgcn_s_wait_asynccnt(0);
#else
    const unsigned int* nf32 = (const unsigned int*)nfeat;
    #pragma unroll 4
    for (int r = 0; r < 32; ++r) {
      int s = s_snd[wave][r];                       // uniform broadcast from LDS
      s_rows[wave][r * 32 + lane] = nf32[s * 32 + lane];  // 128B coalesced row
    }
#endif
    {                                               // splice edge features
      int e = base + lane;
      float4 fv = *(const float4*)(ef + e * 4);
      _Float16* rowh = (_Float16*)&s_rows[wave][lane * 32];
      rowh[34] = (_Float16)fv.x; rowh[35] = (_Float16)fv.y;
      rowh[36] = (_Float16)fv.z; rowh[37] = (_Float16)fv.w;
    }
  }
  __syncthreads();

  if (!valid) return;

  const int row = lane & 15;
  const int b0  = (lane < 16) ? 0 : 8;

  #pragma unroll
  for (int sub = 0; sub < 2; ++sub) {
    Frag16 a[2];
    {
      const uint4* rp = (const uint4*)&s_rows[wave][(sub * 16 + row) * 32];
      #pragma unroll
      for (int kc = 0; kc < 2; ++kc) {
        int q0 = (kc * 16 + (b0 >> 1)) >> 2;   // uint4 index of K segment 0
        a[kc].q[0] = rp[q0];                   // halves K = kc*32 + b0 .. +7
        a[kc].q[1] = rp[q0 + 2];               // halves K = kc*32 + b0+16 .. +23
      }
    }

    int rcvL[8]; float mskL[8];
    #pragma unroll
    for (int i = 0; i < 8; ++i) {
      rcvL[i] = s_rcv[wave][sub * 16 + i + b0];
      mskL[i] = s_msk[wave][sub * 16 + i + b0];
    }

    #pragma unroll
    for (int ct = 0; ct < 2; ++ct) {
      v8f c = {};
      c = __builtin_amdgcn_wmma_f32_16x16x32_f16(false, a[0].v, false, bfrag[ct][0].v,
                                                 (short)0, c, false, false);
      c = __builtin_amdgcn_wmma_f32_16x16x32_f16(false, a[1].v, false, bfrag[ct][1].v,
                                                 (short)0, c, false, false);
      int col = (lane & 15) + ct * 16;
      float bias = bmsg[col];
      #pragma unroll
      for (int i = 0; i < 8; ++i) {
        float v = (c[i] + bias) * mskL[i];
        atomicAdd(&agg[rcvL[i] * 32 + col], v);
      }
    }
  }
}

// ---------------------------------------------------------------------------
// h = relu(agg) in place; pooled[c] += sum over nodes (block-reduced in LDS)
// ---------------------------------------------------------------------------
__global__ void relu_pool_kernel(float* __restrict__ agg,
                                 float* __restrict__ pooled, int N) {
  __shared__ float acc[32];
  int tid = threadIdx.x;
  if (tid < 32) acc[tid] = 0.0f;
  __syncthreads();
  int idx = blockIdx.x * blockDim.x + tid;
  float v = 0.0f;
  if (idx < N * 32) {
    v = agg[idx];
    v = v > 0.0f ? v : 0.0f;
    agg[idx] = v;
  }
  atomicAdd(&acc[tid & 31], v);
  __syncthreads();
  if (tid < 32) atomicAdd(&pooled[tid], acc[tid]);
}

// ---------------------------------------------------------------------------
// g = relu([g, pooled/N] @ W_g + b_g) ; zero pooled for next layer. 32 threads.
// ---------------------------------------------------------------------------
__global__ void global_update_kernel(const float* __restrict__ Wg,
                                     const float* __restrict__ bg,
                                     float* __restrict__ pooled,
                                     float* __restrict__ g, float invN) {
  int c = threadIdx.x;
  float acc = bg[c];
  for (int k = 0; k < 32; ++k) acc += g[k] * Wg[k * 32 + c];
  for (int k = 0; k < 32; ++k) acc += (pooled[k] * invN) * Wg[(32 + k) * 32 + c];
  __syncthreads();
  g[c] = acc > 0.0f ? acc : 0.0f;
  pooled[c] = 0.0f;
}

// ---------------------------------------------------------------------------
// Node update via WMMA: h_new = relu([relu_agg | g] @ W_n + b_n)
// then V += h_new @ W_out + b_out (cross-lane shfl reduction),
// write h_new & V into nfeat (f16), zero agg for next layer.
// ---------------------------------------------------------------------------
__global__ void __launch_bounds__(256) node_update_kernel(
    float* __restrict__ agg, const float* __restrict__ g,
    const _Float16* __restrict__ wpk,   // layer slice W_n packed
    const float* __restrict__ bn, const float* __restrict__ Wout,
    const float* __restrict__ bout,
    _Float16* __restrict__ nfeat, float* __restrict__ V, int ntiles)
{
  const int wave = threadIdx.x >> 5;
  const int lane = threadIdx.x & 31;
  const int tile = blockIdx.x * 8 + wave;
  if (tile >= ntiles) return;
  const int rbase = tile * 16;
  const int b0 = (lane < 16) ? 0 : 8;
  const int row = rbase + (lane & 15);

  Frag16 bfrag[2][2];
  {
    const uint4* wq = (const uint4*)wpk;
    #pragma unroll
    for (int ct = 0; ct < 2; ++ct)
      #pragma unroll
      for (int kc = 0; kc < 2; ++kc) {
        int qi = ((ct * 2 + kc) * 32 + lane) * 2;
        bfrag[ct][kc].q[0] = wq[qi];
        bfrag[ct][kc].q[1] = wq[qi + 1];
      }
  }

  Frag16 a[2];
  {
    const float* hp = agg + row * 32;   // relu'd by relu_pool_kernel
    #pragma unroll
    for (int j = 0; j < 8; ++j) a[0].v[j]     = (_Float16)hp[b0 + j];
    #pragma unroll
    for (int j = 0; j < 8; ++j) a[0].v[8 + j] = (_Float16)hp[b0 + 16 + j];
    #pragma unroll
    for (int j = 0; j < 8; ++j) a[1].v[j]     = (_Float16)g[b0 + j];
    #pragma unroll
    for (int j = 0; j < 8; ++j) a[1].v[8 + j] = (_Float16)g[b0 + 16 + j];
  }

  float pv0[8], pv1[8];
  #pragma unroll
  for (int i = 0; i < 8; ++i) { pv0[i] = 0.0f; pv1[i] = 0.0f; }

  #pragma unroll
  for (int ct = 0; ct < 2; ++ct) {
    v8f c = {};
    c = __builtin_amdgcn_wmma_f32_16x16x32_f16(false, a[0].v, false, bfrag[ct][0].v,
                                               (short)0, c, false, false);
    c = __builtin_amdgcn_wmma_f32_16x16x32_f16(false, a[1].v, false, bfrag[ct][1].v,
                                               (short)0, c, false, false);
    int col = (lane & 15) + ct * 16;
    float bb = bn[col];
    float w0 = Wout[col * 2], w1 = Wout[col * 2 + 1];
    #pragma unroll
    for (int i = 0; i < 8; ++i) {
      float h = c[i] + bb;
      h = h > 0.0f ? h : 0.0f;
      int node = rbase + i + b0;
      nfeat[node * 64 + 2 + col] = (_Float16)h;
      agg[node * 32 + col] = 0.0f;     // ready for next layer's scatter
      pv0[i] += h * w0;
      pv1[i] += h * w1;
    }
  }

  // reduce over 16 lanes within each half-wave (cols 0..31 -> per-row dV)
  #pragma unroll
  for (int off = 1; off < 16; off <<= 1) {
    #pragma unroll
    for (int i = 0; i < 8; ++i) {
      pv0[i] += __shfl_xor(pv0[i], off, 32);
      pv1[i] += __shfl_xor(pv1[i], off, 32);
    }
  }
  if ((lane & 15) == 0) {              // lane 0 -> rows 0..7, lane 16 -> rows 8..15
    float bo0 = bout[0], bo1 = bout[1];
    #pragma unroll
    for (int i = 0; i < 8; ++i) {
      int node = rbase + i + b0;
      float v0 = V[node * 2]     + pv0[i] + bo0;
      float v1 = V[node * 2 + 1] + pv1[i] + bo1;
      V[node * 2]     = v0;
      V[node * 2 + 1] = v1;
      nfeat[node * 64]     = (_Float16)v0;
      nfeat[node * 64 + 1] = (_Float16)v1;
    }
  }
}

// ---------------------------------------------------------------------------
extern "C" void kernel_launch(void* const* d_in, const int* in_sizes, int n_in,
                              void* d_out, int out_size, void* d_ws, size_t ws_size,
                              hipStream_t stream) {
  const float* P         = (const float*)d_in[0];
  const int*   senders   = (const int*)d_in[1];
  const int*   receivers = (const int*)d_in[2];
  const float* ef        = (const float*)d_in[3];
  const float* emask     = (const float*)d_in[4];
  const float* W_in      = (const float*)d_in[5];
  const float* b_in      = (const float*)d_in[6];
  const float* W_msg     = (const float*)d_in[7];
  const float* b_msg     = (const float*)d_in[8];
  const float* W_g       = (const float*)d_in[9];
  const float* b_g       = (const float*)d_in[10];
  const float* W_n       = (const float*)d_in[11];
  const float* b_n       = (const float*)d_in[12];
  const float* W_out     = (const float*)d_in[13];
  const float* b_out     = (const float*)d_in[14];
  (void)n_in; (void)out_size; (void)ws_size;

  const int N = in_sizes[0] / 2;
  const int E = in_sizes[1];
  float* V = (float*)d_out;

  char* ws = (char*)d_ws;
  size_t off = 0;
  auto carve = [&](size_t bytes) -> void* {
    void* p = ws + off;
    off = (off + bytes + 255) & ~(size_t)255;
    return p;
  };
  _Float16* nfeat   = (_Float16*)carve((size_t)N * 64 * sizeof(_Float16));
  float*    agg     = (float*)   carve((size_t)N * 32 * sizeof(float));
  float*    pooled  = (float*)   carve(128);
  float*    g       = (float*)   carve(128);
  _Float16* wmsg_pk = (_Float16*)carve(3 * 4 * 512 * sizeof(_Float16));
  _Float16* wn_pk   = (_Float16*)carve(3 * 4 * 512 * sizeof(_Float16));

  prep_pack_kernel<<<1, 256, 0, stream>>>(W_msg, W_n, wmsg_pk, wn_pk);

  const int nc_blocks = (N * 32 + 255) / 256;
  init_kernel<<<nc_blocks, 256, 0, stream>>>(P, W_in, b_in, nfeat, agg, pooled, g, V, N);

  const int etiles  = E / 32;                 // 100,000 (32 edges / wave)
  const int eblocks = (etiles + 7) / 8;
  const int ntiles  = N / 16;                 // 6,250
  const int nblocks = (ntiles + 7) / 8;
  const float invN  = 1.0f / (float)N;

  for (int l = 0; l < 3; ++l) {
    edge_msg_kernel<<<eblocks, 256, 0, stream>>>(
        nfeat, senders, receivers, ef, emask,
        wmsg_pk + l * 2048, b_msg + l * 32, agg, etiles);
    relu_pool_kernel<<<nc_blocks, 256, 0, stream>>>(agg, pooled, N);
    global_update_kernel<<<1, 32, 0, stream>>>(
        W_g + l * 64 * 32, b_g + l * 32, pooled, g, invN);
    node_update_kernel<<<nblocks, 256, 0, stream>>>(
        agg, g, wn_pk + l * 2048, b_n + l * 32,
        W_out + l * 64, b_out + l * 2, nfeat, V, ntiles);
  }
}
